// OptimizedGAT_56702158241979
// MI455X (gfx1250) — compile-verified, matrix operand order checked by reading, not run on hardware
//
#include <hip/hip_runtime.h>
#include <hip/hip_bf16.h>
#include <math.h>

#define N_NODES_C   50000
#define N_EDGES_C   800000
#define IN_CH_C     128
#define HID_C       128
#define HEADS_C     2
#define F1_C        (HEADS_C * HID_C)   // 256
#define OUT2_C      64
#define NUM_CLASSES_C 2
#define NUM_GRAPHS_C  64
#define NEG_SLOPE_C 0.2f

typedef __attribute__((ext_vector_type(16))) __bf16 v16bf;
typedef __attribute__((ext_vector_type(8)))  __bf16 v8bf;
typedef __attribute__((ext_vector_type(8)))  float  v8f;

// ---------------------------------------------------------------------------
// Utility kernels
// ---------------------------------------------------------------------------
__global__ void gat_fill_f32(float* __restrict__ p, float v, long long n) {
    long long i = (long long)blockIdx.x * blockDim.x + threadIdx.x;
    if (i < n) p[i] = v;
}

__global__ void gat_f32_to_bf16(const float* __restrict__ in, __bf16* __restrict__ out, long long n) {
    long long i = (long long)blockIdx.x * blockDim.x + threadIdx.x;
    if (i < n) out[i] = (__bf16)in[i];
}

// W is [K,N] row-major fp32 -> Wt is [N,K] row-major bf16
__global__ void gat_transpose_bf16(const float* __restrict__ W, __bf16* __restrict__ Wt, int K, int N) {
    int i = blockIdx.x * blockDim.x + threadIdx.x;
    if (i >= K * N) return;
    int k = i / N, n = i % N;
    Wt[(size_t)n * K + k] = (__bf16)W[(size_t)k * N + n];
}

__device__ inline void gat_atomic_max_f32(float* addr, float val) {
    int* ia = (int*)addr;
    int old = __float_as_int(*addr);
    while (__int_as_float(old) < val) {
        int assumed = old;
        old = atomicCAS(ia, assumed, __float_as_int(val));
        if (old == assumed) break;
    }
}

// ---------------------------------------------------------------------------
// WMMA GEMM: C[M,N] = A[M,K](bf16) * Bt[N,K](bf16, pre-transposed), fp32 out.
// One wave32 per 16x16 output tile; K stepped by 32 per v_wmma_f32_16x16x32_bf16.
// A-fragment layout (16-bit, 16x32): lanes 0-15 hold M=lane, K in {0..7,16..23};
// lanes 16-31 hold same M, K in {8..15,24..31}. B mirrors with N per lane.
// C/D layout: VGPR r -> M = r + 8*(lane>=16), N = lane&15.
// ---------------------------------------------------------------------------
template <int N, int K>
__global__ void __launch_bounds__(256)
gat_wmma_gemm_bf16(const __bf16* __restrict__ A, const __bf16* __restrict__ Bt,
                   float* __restrict__ C, int M) {
    const int lane = threadIdx.x & 31;
    const int gw   = (int)((blockIdx.x * blockDim.x + threadIdx.x) >> 5);
    const int tilesN = N / 16;
    const int tilesM = M / 16;
    if (gw >= tilesM * tilesN) return;          // wave-uniform guard (EXEC stays all-1 for WMMA)
    const int tm = gw / tilesN;
    const int tn = gw % tilesN;
    const int li = lane & 15;                   // M (for A) / N (for B) within tile
    const int kb = (lane >> 4) * 8;             // K sub-block base per lane half

    const __bf16* arow = A  + (size_t)(tm * 16 + li) * K;
    const __bf16* brow = Bt + (size_t)(tn * 16 + li) * K;

    v8f acc = {};
    for (int k0 = 0; k0 < K; k0 += 32) {
        __builtin_prefetch(arow + k0 + 64, 0, 0);   // global_prefetch_b8 next K blocks
        v8bf alo = *(const v8bf*)(arow + k0 + kb);
        v8bf ahi = *(const v8bf*)(arow + k0 + kb + 16);
        v8bf blo = *(const v8bf*)(brow + k0 + kb);
        v8bf bhi = *(const v8bf*)(brow + k0 + kb + 16);
        v16bf a, b;
#pragma unroll
        for (int i = 0; i < 8; ++i) {
            a[i] = alo[i]; a[i + 8] = ahi[i];
            b[i] = blo[i]; b[i + 8] = bhi[i];
        }
        acc = __builtin_amdgcn_wmma_f32_16x16x32_bf16(
            /*neg_a=*/false, a, /*neg_b=*/false, b,
            /*c_mod=*/(short)0, acc, /*reuse_a=*/false, /*reuse_b=*/false);
    }

    float* crow = C + (size_t)(tm * 16 + (lane >> 4) * 8) * N + tn * 16 + li;
#pragma unroll
    for (int r = 0; r < 8; ++r) crow[(size_t)r * N] = acc[r];
}

// ---------------------------------------------------------------------------
// Per-node attention dots: as[n,h] = dot(h[n,h,:], att_s[h]); same for ad.
// ---------------------------------------------------------------------------
__global__ void gat_att_scores(const float* __restrict__ h, const float* __restrict__ att_s,
                               const float* __restrict__ att_d, float* __restrict__ as,
                               float* __restrict__ ad, int Nn, int H, int C) {
    int i = blockIdx.x * blockDim.x + threadIdx.x;
    if (i >= Nn * H) return;
    int n = i / H, hd = i % H;
    const float* hp = h + (size_t)n * H * C + (size_t)hd * C;
    const float* sp = att_s + (size_t)hd * C;
    const float* dp = att_d + (size_t)hd * C;
    float ss = 0.f, sd = 0.f;
    for (int c = 0; c < C; ++c) { float v = hp[c]; ss += v * sp[c]; sd += v * dp[c]; }
    as[i] = ss; ad[i] = sd;
}

// ---------------------------------------------------------------------------
// Edge softmax passes (segment_max / segment_sum over dst)
// ---------------------------------------------------------------------------
__global__ void gat_edge_max(const int* __restrict__ src, const int* __restrict__ dst,
                             const float* __restrict__ as, const float* __restrict__ ad,
                             float* __restrict__ m, int H) {
    long long idx = (long long)blockIdx.x * blockDim.x + threadIdx.x;
    if (idx >= (long long)N_EDGES_C * H) return;
    int hd = (int)(idx % H); int e = (int)(idx / H);
    int s = src[e], d = dst[e];
    float v = as[(size_t)s * H + hd] + ad[(size_t)d * H + hd];
    v = v > 0.f ? v : NEG_SLOPE_C * v;
    gat_atomic_max_f32(&m[(size_t)d * H + hd], v);
}

__global__ void gat_edge_exp(const int* __restrict__ src, const int* __restrict__ dst,
                             const float* __restrict__ as, const float* __restrict__ ad,
                             const float* __restrict__ m, float* __restrict__ ex,
                             float* __restrict__ den, int H) {
    long long idx = (long long)blockIdx.x * blockDim.x + threadIdx.x;
    if (idx >= (long long)N_EDGES_C * H) return;
    int hd = (int)(idx % H); int e = (int)(idx / H);
    int s = src[e], d = dst[e];
    float v = as[(size_t)s * H + hd] + ad[(size_t)d * H + hd];
    v = v > 0.f ? v : NEG_SLOPE_C * v;
    float xv = __expf(v - m[(size_t)d * H + hd]);
    ex[idx] = xv;
    atomicAdd(&den[(size_t)d * H + hd], xv);
}

// agg[dst, h, :] += h[src, h, :] * alpha   (float4 per thread)
__global__ void gat_edge_agg(const int* __restrict__ src, const int* __restrict__ dst,
                             const float* __restrict__ ex, const float* __restrict__ den,
                             const float* __restrict__ h, float* __restrict__ agg,
                             int H, int C4) {
    long long idx = (long long)blockIdx.x * blockDim.x + threadIdx.x;
    long long total = (long long)N_EDGES_C * H * C4;
    if (idx >= total) return;
    int cv = (int)(idx % C4);
    long long t = idx / C4;
    int hd = (int)(t % H);
    int e  = (int)(t / H);
    int s = src[e], d = dst[e];
    int F = H * C4 * 4;
    float alpha = ex[(size_t)e * H + hd] / (den[(size_t)d * H + hd] + 1e-16f);
    const float4 hv = *(const float4*)(h + (size_t)s * F + (size_t)hd * C4 * 4 + cv * 4);
    float* out = agg + (size_t)d * F + (size_t)hd * C4 * 4 + cv * 4;
    atomicAdd(out + 0, hv.x * alpha);
    atomicAdd(out + 1, hv.y * alpha);
    atomicAdd(out + 2, hv.z * alpha);
    atomicAdd(out + 3, hv.w * alpha);
}

// ---------------------------------------------------------------------------
// ELU epilogues
// ---------------------------------------------------------------------------
__global__ void gat_elu_bias_bf16(const float* __restrict__ agg, const float* __restrict__ bias,
                                  __bf16* __restrict__ out, long long n, int F) {
    long long i = (long long)blockIdx.x * blockDim.x + threadIdx.x;
    if (i >= n) return;
    float v = agg[i] + bias[(int)(i % F)];
    v = v > 0.f ? v : expm1f(v);
    out[i] = (__bf16)v;
}

__global__ void gat_elu_bias_f32(const float* __restrict__ agg, const float* __restrict__ bias,
                                 float* __restrict__ out, long long n, int F) {
    long long i = (long long)blockIdx.x * blockDim.x + threadIdx.x;
    if (i >= n) return;
    float v = agg[i] + bias[(int)(i % F)];
    out[i] = v > 0.f ? v : expm1f(v);
}

// ---------------------------------------------------------------------------
// Global mean pool + classifier
// ---------------------------------------------------------------------------
__global__ void gat_pool(const float* __restrict__ h, const int* __restrict__ batch,
                         float* __restrict__ sums, float* __restrict__ counts) {
    long long i = (long long)blockIdx.x * blockDim.x + threadIdx.x;
    if (i >= (long long)N_NODES_C * OUT2_C) return;
    int n = (int)(i / OUT2_C), c = (int)(i % OUT2_C);
    int g = batch[n];
    atomicAdd(&sums[(size_t)g * OUT2_C + c], h[i]);
    if (c == 0) atomicAdd(&counts[g], 1.0f);
}

__global__ void gat_classify(const float* __restrict__ sums, const float* __restrict__ counts,
                             const float* __restrict__ Wc, const float* __restrict__ bc,
                             float* __restrict__ out) {
    int i = blockIdx.x * blockDim.x + threadIdx.x;
    if (i >= NUM_GRAPHS_C * NUM_CLASSES_C) return;
    int g = i / NUM_CLASSES_C, cls = i % NUM_CLASSES_C;
    float cnt = counts[g]; cnt = cnt > 1.f ? cnt : 1.f;
    float acc = 0.f;
    for (int c = 0; c < OUT2_C; ++c)
        acc += (sums[(size_t)g * OUT2_C + c] / cnt) * Wc[c * NUM_CLASSES_C + cls];
    out[i] = acc + bc[cls];
}

// ---------------------------------------------------------------------------
// Launch
// ---------------------------------------------------------------------------
static inline int gblocks(long long n, int b) { return (int)((n + b - 1) / b); }

extern "C" void kernel_launch(void* const* d_in, const int* in_sizes, int n_in,
                              void* d_out, int out_size, void* d_ws, size_t ws_size,
                              hipStream_t stream) {
    (void)in_sizes; (void)n_in; (void)out_size; (void)ws_size;
    const float* x        = (const float*)d_in[0];
    const int*   ei       = (const int*)d_in[1];
    const int*   batch    = (const int*)d_in[2];
    const float* W1       = (const float*)d_in[3];
    const float* att_src1 = (const float*)d_in[4];
    const float* att_dst1 = (const float*)d_in[5];
    const float* b1       = (const float*)d_in[6];
    const float* W2       = (const float*)d_in[7];
    const float* att_src2 = (const float*)d_in[8];
    const float* att_dst2 = (const float*)d_in[9];
    const float* b2       = (const float*)d_in[10];
    const float* Wc       = (const float*)d_in[11];
    const float* bc       = (const float*)d_in[12];
    float* out = (float*)d_out;

    const int* src = ei;
    const int* dst = ei + N_EDGES_C;

    // ---- workspace carve-out (256B aligned) ----
    char* ws = (char*)d_ws;
    size_t off = 0;
    auto alloc = [&](size_t bytes) -> void* {
        void* p = ws + off;
        off = (off + bytes + 255) & ~(size_t)255;
        return p;
    };
    __bf16* xb   = (__bf16*)alloc((size_t)N_NODES_C * IN_CH_C * 2);
    __bf16* w1t  = (__bf16*)alloc((size_t)F1_C * IN_CH_C * 2);
    __bf16* w2t  = (__bf16*)alloc((size_t)OUT2_C * F1_C * 2);
    float*  h1   = (float*) alloc((size_t)N_NODES_C * F1_C * 4);
    float*  as1  = (float*) alloc((size_t)N_NODES_C * HEADS_C * 4);
    float*  ad1  = (float*) alloc((size_t)N_NODES_C * HEADS_C * 4);
    float*  m1   = (float*) alloc((size_t)N_NODES_C * HEADS_C * 4);
    float*  den1 = (float*) alloc((size_t)N_NODES_C * HEADS_C * 4);
    float*  ex1  = (float*) alloc((size_t)N_EDGES_C * HEADS_C * 4);
    float*  agg1 = (float*) alloc((size_t)N_NODES_C * F1_C * 4);
    float*  h2   = (float*) alloc((size_t)N_NODES_C * OUT2_C * 4);
    float*  as2  = (float*) alloc((size_t)N_NODES_C * 4);
    float*  ad2  = (float*) alloc((size_t)N_NODES_C * 4);
    float*  m2   = (float*) alloc((size_t)N_NODES_C * 4);
    float*  den2 = (float*) alloc((size_t)N_NODES_C * 4);
    float*  ex2  = (float*) alloc((size_t)N_EDGES_C * 4);
    float*  agg2 = (float*) alloc((size_t)N_NODES_C * OUT2_C * 4);
    float*  psum = (float*) alloc((size_t)NUM_GRAPHS_C * OUT2_C * 4);
    float*  pcnt = (float*) alloc((size_t)NUM_GRAPHS_C * 4);
    // overlays onto dead regions (h1 dead after agg1, h2 dead after agg2)
    __bf16* h1e = (__bf16*)h1;   // bf16 [N, 256]
    float*  h2e = h2;            // f32  [N, 64]

    const int B = 256;

    // ---- re-init accumulators every call (atomics below) ----
    gat_fill_f32<<<gblocks((long long)N_NODES_C * F1_C, B), B, 0, stream>>>(agg1, 0.f, (long long)N_NODES_C * F1_C);
    gat_fill_f32<<<gblocks((long long)N_NODES_C * OUT2_C, B), B, 0, stream>>>(agg2, 0.f, (long long)N_NODES_C * OUT2_C);
    gat_fill_f32<<<gblocks((long long)N_NODES_C * HEADS_C, B), B, 0, stream>>>(den1, 0.f, (long long)N_NODES_C * HEADS_C);
    gat_fill_f32<<<gblocks((long long)N_NODES_C, B), B, 0, stream>>>(den2, 0.f, (long long)N_NODES_C);
    gat_fill_f32<<<gblocks((long long)N_NODES_C * HEADS_C, B), B, 0, stream>>>(m1, -INFINITY, (long long)N_NODES_C * HEADS_C);
    gat_fill_f32<<<gblocks((long long)N_NODES_C, B), B, 0, stream>>>(m2, -INFINITY, (long long)N_NODES_C);
    gat_fill_f32<<<gblocks((long long)NUM_GRAPHS_C * OUT2_C, B), B, 0, stream>>>(psum, 0.f, (long long)NUM_GRAPHS_C * OUT2_C);
    gat_fill_f32<<<gblocks((long long)NUM_GRAPHS_C, B), B, 0, stream>>>(pcnt, 0.f, (long long)NUM_GRAPHS_C);

    // ---- precision conversion / weight transpose ----
    gat_f32_to_bf16<<<gblocks((long long)N_NODES_C * IN_CH_C, B), B, 0, stream>>>(x, xb, (long long)N_NODES_C * IN_CH_C);
    gat_transpose_bf16<<<gblocks((long long)IN_CH_C * F1_C, B), B, 0, stream>>>(W1, w1t, IN_CH_C, F1_C);
    gat_transpose_bf16<<<gblocks((long long)F1_C * OUT2_C, B), B, 0, stream>>>(W2, w2t, F1_C, OUT2_C);

    // ---- layer 1 ----
    {
        long long waves = (long long)(N_NODES_C / 16) * (F1_C / 16);   // 50000
        gat_wmma_gemm_bf16<F1_C, IN_CH_C><<<gblocks(waves * 32, B), B, 0, stream>>>(xb, w1t, h1, N_NODES_C);
    }
    gat_att_scores<<<gblocks((long long)N_NODES_C * HEADS_C, B), B, 0, stream>>>(h1, att_src1, att_dst1, as1, ad1, N_NODES_C, HEADS_C, HID_C);
    gat_edge_max<<<gblocks((long long)N_EDGES_C * HEADS_C, B), B, 0, stream>>>(src, dst, as1, ad1, m1, HEADS_C);
    gat_edge_exp<<<gblocks((long long)N_EDGES_C * HEADS_C, B), B, 0, stream>>>(src, dst, as1, ad1, m1, ex1, den1, HEADS_C);
    gat_edge_agg<<<gblocks((long long)N_EDGES_C * HEADS_C * (HID_C / 4), B), B, 0, stream>>>(src, dst, ex1, den1, h1, agg1, HEADS_C, HID_C / 4);
    gat_elu_bias_bf16<<<gblocks((long long)N_NODES_C * F1_C, B), B, 0, stream>>>(agg1, b1, h1e, (long long)N_NODES_C * F1_C, F1_C);

    // ---- layer 2 ----
    {
        long long waves = (long long)(N_NODES_C / 16) * (OUT2_C / 16); // 12500
        gat_wmma_gemm_bf16<OUT2_C, F1_C><<<gblocks(waves * 32, B), B, 0, stream>>>(h1e, w2t, h2, N_NODES_C);
    }
    gat_att_scores<<<gblocks((long long)N_NODES_C, B), B, 0, stream>>>(h2, att_src2, att_dst2, as2, ad2, N_NODES_C, 1, OUT2_C);
    gat_edge_max<<<gblocks((long long)N_EDGES_C, B), B, 0, stream>>>(src, dst, as2, ad2, m2, 1);
    gat_edge_exp<<<gblocks((long long)N_EDGES_C, B), B, 0, stream>>>(src, dst, as2, ad2, m2, ex2, den2, 1);
    gat_edge_agg<<<gblocks((long long)N_EDGES_C * (OUT2_C / 4), B), B, 0, stream>>>(src, dst, ex2, den2, h2, agg2, 1, OUT2_C / 4);
    gat_elu_bias_f32<<<gblocks((long long)N_NODES_C * OUT2_C, B), B, 0, stream>>>(agg2, b2, h2e, (long long)N_NODES_C * OUT2_C, OUT2_C);

    // ---- pool + classify ----
    gat_pool<<<gblocks((long long)N_NODES_C * OUT2_C, B), B, 0, stream>>>(h2e, batch, psum, pcnt);
    gat_classify<<<1, 128, 0, stream>>>(psum, pcnt, Wc, bc, out);
}
